// QuantizedDecoder_51316269252995
// MI455X (gfx1250) — compile-verified, matrix operand order checked by reading, not run on hardware
//
#include <hip/hip_runtime.h>
#include <hip/hip_bf16.h>

typedef __attribute__((ext_vector_type(16))) _Float16 v16h;
typedef __attribute__((ext_vector_type(8)))  _Float16 v8h;
typedef __attribute__((ext_vector_type(8)))  float    v8f;

#define N_ROWS   131072
#define IN_DIM   32
#define HID      512
#define LAT      16
#define NEMB     1024
#define OUT3A    18      // actual cols of layer 3 (LAT + 2)
#define OUT3P    32      // padded to 2 n-tiles
#define BM       64      // rows per block
#define THREADS  256     // 8 wave32s

// ---- d_out layout (floats), outputs concatenated flat in return order ----
#define ZQ_OFF        0
#define SCALER_OFF    (N_ROWS * LAT)               // 2097152
#define REDSHIFT_OFF  (SCALER_OFF + N_ROWS)        // 2228224
#define LOSS_OFF      (REDSHIFT_OFF + N_ROWS)      // 2359296
#define IDS_OFF       (LOSS_OFF + 1)               // 2359297

// ---- d_ws layout (offsets in _Float16 elements) ----
#define W0P_OFF   0
#define W0P_SZ    (IN_DIM * HID)        // 16384
#define W1P_OFF   (W0P_OFF + W0P_SZ)
#define W1P_SZ    (HID * HID)           // 262144
#define W2P_OFF   (W1P_OFF + W1P_SZ)
#define W2P_SZ    (HID * HID)
#define W3P_OFF   (W2P_OFF + W2P_SZ)
#define W3P_SZ    (HID * OUT3P)         // 16384
#define CBP_OFF   (W3P_OFF + W3P_SZ)
#define CBP_SZ    (32 * NEMB)           // 32768  (K padded 16->32)
#define CB2_OFF_H (CBP_OFF + CBP_SZ)    // float array starts here (even -> 4B aligned)

// ---------------------------------------------------------------------------
// Pre-pack a row-major (Kact x Nact) f32 matrix into f16 WMMA-B fragment
// order for v_wmma_f32_16x16x32_f16: tile t = nt*(K/32)+kt; within a tile,
// lane l holds 16 contiguous halves = its column's K-slice in VGPR order.
// ---------------------------------------------------------------------------
__global__ void pack_b_kernel(const float* __restrict__ src, _Float16* __restrict__ dst,
                              int K, int Kact, int Ntot, int Nact)
{
    int idx   = blockIdx.x * blockDim.x + threadIdx.x;
    int total = K * Ntot;
    if (idx >= total) return;
    int i    = idx & 15;          // half index within lane's fragment
    int lane = (idx >> 4) & 31;
    int t    = idx >> 9;          // tile index
    int KT   = K >> 5;
    int kt   = t % KT;
    int nt   = t / KT;
    int v = i >> 1, h = i & 1;
    int n = nt * 16 + (lane & 15);
    int k = kt * 32 + ((lane >> 4) * 8)
          + ((v < 4) ? (v * 2 + h) : (16 + (v - 4) * 2 + h));
    float val = (k < Kact && n < Nact) ? src[k * Nact + n] : 0.0f;
    dst[idx] = (_Float16)val;
}

// 0.5 * ||codebook column n||^2 (for argmin -> argmax trick)
__global__ void cb2_kernel(const float* __restrict__ cb, float* __restrict__ cb2)
{
    int n = blockIdx.x * blockDim.x + threadIdx.x;
    if (n < NEMB) {
        float s = 0.0f;
        for (int l = 0; l < LAT; ++l) { float c = cb[l * NEMB + n]; s += c * c; }
        cb2[n] = 0.5f * s;
    }
}

// d_out is poisoned once and never re-zeroed between replays; the loss cell is
// accumulated with atomics, so zero it at the start of every launch.
__global__ void zero_loss_kernel(float* __restrict__ out)
{
    if (blockIdx.x == 0 && threadIdx.x == 0) out[LOSS_OFF] = 0.0f;
}

// ---------------------------------------------------------------------------
// Fragment loaders
// ---------------------------------------------------------------------------
// A fragment (16x32 f16) from a row-major f16 LDS buffer. Lane l (0..15) holds
// row m0+l, K-halves {kb..kb+7, kb+16..kb+23}; lane l+16 holds the +8 slices.
__device__ __forceinline__ v16h load_a_lds(const _Float16* base, int stride,
                                           int row0, int kbase, int lane)
{
    int r  = lane & 15;
    int hs = lane >> 4;
    const _Float16* p = base + (row0 + r) * stride + kbase + hs * 8;
    v8h lo = *(const v8h*)(p);
    v8h hi = *(const v8h*)(p + 16);
    v16h a;
#pragma unroll
    for (int i = 0; i < 8; ++i) { a[i] = lo[i]; a[i + 8] = hi[i]; }
    return a;
}

// B fragment (32x16 f16) from the pre-packed global buffer: 16 contiguous
// halves per lane -> two global_load_b128.
__device__ __forceinline__ v16h load_b_glob(const _Float16* wp, int t, int lane)
{
    const _Float16* p = wp + (((long long)t * 32 + lane) << 4);
    v8h lo = *(const v8h*)(p);
    v8h hi = *(const v8h*)(p + 8);
    v16h b;
#pragma unroll
    for (int i = 0; i < 8; ++i) { b[i] = lo[i]; b[i + 8] = hi[i]; }
    return b;
}

__device__ __forceinline__ v8f wmma_f16(v16h a, v16h b, v8f c)
{
    return __builtin_amdgcn_wmma_f32_16x16x32_f16(false, a, false, b,
                                                  (short)0, c, false, false);
}

// ---------------------------------------------------------------------------
// One GEMM layer with relu for this block's 64-row tile, 4x4 register
// blocking: each wave owns ALL 4 m-tiles and 4 n-tiles {w*4 .. w*4+3}.
// Per k-step: 4 A frags (8 ds_load_b128) + 4 B frags (8 global_load_b128)
// feed 16 WMMAs; every weight byte is fetched exactly once per block.
// ---------------------------------------------------------------------------
template<int KSTEPS, int ASTRIDE>
__device__ __forceinline__ void gemm_relu_4x4(const _Float16* __restrict__ hsrc,
                                              const _Float16* __restrict__ Wp,
                                              const float* __restrict__ bias,
                                              _Float16* __restrict__ hdst,
                                              int w, int lane)
{
    const int ln15 = lane & 15;
    const int hs   = lane >> 4;
    v8f acc[4][4];
#pragma unroll
    for (int m = 0; m < 4; ++m)
#pragma unroll
        for (int j = 0; j < 4; ++j)
#pragma unroll
            for (int k = 0; k < 8; ++k) acc[m][j][k] = 0.0f;

    for (int kt = 0; kt < KSTEPS; ++kt) {
        v16h a[4];
#pragma unroll
        for (int m = 0; m < 4; ++m)
            a[m] = load_a_lds(hsrc, ASTRIDE, m * 16, kt * 32, lane);
#pragma unroll
        for (int j = 0; j < 4; ++j) {
            v16h b = load_b_glob(Wp, (w * 4 + j) * KSTEPS + kt, lane);
#pragma unroll
            for (int m = 0; m < 4; ++m)
                acc[m][j] = wmma_f16(a[m], b, acc[m][j]);
        }
    }
#pragma unroll
    for (int j = 0; j < 4; ++j) {
        int nn = (w * 4 + j) * 16 + ln15;
        float bv = bias[nn];
#pragma unroll
        for (int m = 0; m < 4; ++m)
#pragma unroll
            for (int r = 0; r < 8; ++r) {
                float v = acc[m][j][r] + bv;
                v = v > 0.0f ? v : 0.0f;
                hdst[(m * 16 + r + hs * 8) * HID + nn] = (_Float16)v;
            }
    }
}

// ---------------------------------------------------------------------------
// Fused decoder + VQ kernel: one block = 64 rows, activations stay in LDS.
// ---------------------------------------------------------------------------
__global__ void __launch_bounds__(THREADS)
fused_decoder_vq(const float* __restrict__ z,
                 const _Float16* __restrict__ W0p, const float* __restrict__ b0,
                 const _Float16* __restrict__ W1p, const float* __restrict__ b1,
                 const _Float16* __restrict__ W2p, const float* __restrict__ b2,
                 const _Float16* __restrict__ W3p, const float* __restrict__ b3,
                 const _Float16* __restrict__ CBp, const float* __restrict__ cb2,
                 const float* __restrict__ codebook,
                 float* __restrict__ out)
{
    extern __shared__ char smem_raw[];
    _Float16* zb = (_Float16*)smem_raw;            // BM*32  f16 (z, later zl)
    _Float16* hA = zb + BM * 32;                   // BM*512 f16
    _Float16* hB = hA + BM * HID;                  // BM*512 f16
    float* ob    = (float*)(hB + BM * HID);        // BM*32  f32 (layer-3 out)
    float* bestv = ob + BM * OUT3P;                // BM*32  f32 (VQ partials)
    int*   besti = (int*)(bestv + BM * 32);        // BM*32  i32
    float* lossp = (float*)(besti + BM * 32);      // BM     f32

    const int tid  = threadIdx.x;
    const int lane = tid & 31;
    const int w    = tid >> 5;
    const int ln15 = lane & 15;
    const int hs   = lane >> 4;
    const long long rowbase = (long long)blockIdx.x * BM;

    // ---- stage z tile into LDS as f16 ----
    for (int i = tid; i < BM * IN_DIM; i += THREADS)
        zb[i] = (_Float16)z[rowbase * IN_DIM + i];
    __syncthreads();

    // ---- layer 0: hA = relu(z @ W0 + b0), K=32 single step ----
    gemm_relu_4x4<1, IN_DIM>(zb, W0p, b0, hA, w, lane);
    __syncthreads();

    // ---- layers 1 & 2 (ping-pong hA <-> hB) ----
    gemm_relu_4x4<16, HID>(hA, W1p, b1, hB, w, lane);
    __syncthreads();
    gemm_relu_4x4<16, HID>(hB, W2p, b2, hA, w, lane);
    __syncthreads();

    // ---- layer 3: ob(64x32 f32) = hA @ W3pad + b3 ----
    // wave w: m-tile = w&3, n-tile = w>>2 (8 tiles total)
    {
        const int mt = w & 3;
        const int p  = w >> 2;
        const int m0 = mt * 16;
        v8f acc;
#pragma unroll
        for (int k = 0; k < 8; ++k) acc[k] = 0.0f;
        for (int kt = 0; kt < 16; ++kt) {
            v16h a = load_a_lds(hA, HID, m0, kt * 32, lane);
            v16h b = load_b_glob(W3p, p * 16 + kt, lane);  // t = nt*16 + kt
            acc = wmma_f16(a, b, acc);
        }
        int nn = p * 16 + ln15;
        float bv = (nn < OUT3A) ? b3[nn] : 0.0f;
#pragma unroll
        for (int r = 0; r < 8; ++r)
            ob[(m0 + r + hs * 8) * OUT3P + nn] = acc[r] + bv;
    }
    __syncthreads();

    // ---- build zl f16 (K padded 16->32 with zeros) into zb ----
    for (int i = tid; i < BM * 32; i += THREADS) {
        int k = i & 31;
        zb[i] = (_Float16)((k < LAT) ? ob[i] : 0.0f);
    }
    __syncthreads();

    // ---- VQ: scores = zl @ codebook; argmin d == argmax(score - 0.5||c||^2)
    // wave w: m-tile = w&3, n-tiles (w>>2)*32 .. +31 (codebook is L0-resident)
    {
        const int mt = w & 3;
        const int p  = w >> 2;
        const int m0 = mt * 16;
        v16h a = load_a_lds(zb, 32, m0, 0, lane);
        float bv_[8]; int bi_[8];
#pragma unroll
        for (int r = 0; r < 8; ++r) { bv_[r] = -3.0e38f; bi_[r] = 0; }
        for (int j = 0; j < 32; ++j) {
            int nt = p * 32 + j;
            v16h b = load_b_glob(CBp, nt, lane);
            v8f acc;
#pragma unroll
            for (int k = 0; k < 8; ++k) acc[k] = 0.0f;
            acc = wmma_f16(a, b, acc);
            int nn = nt * 16 + ln15;
            float c2 = cb2[nn];
#pragma unroll
            for (int r = 0; r < 8; ++r) {
                float s = acc[r] - c2;
                if (s > bv_[r]) { bv_[r] = s; bi_[r] = nn; }
            }
        }
#pragma unroll
        for (int r = 0; r < 8; ++r) {
            int row  = m0 + r + hs * 8;
            int slot = p * 16 + ln15;
            bestv[row * 32 + slot] = bv_[r];
            besti[row * 32 + slot] = bi_[r];
        }
    }
    __syncthreads();

    // ---- final per-row reduction + outputs ----
    if (tid < BM) {
        int row = tid;
        float best = -3.0e38f; int bidx = 0;
        for (int sl = 0; sl < 32; ++sl) {
            float v = bestv[row * 32 + sl];
            if (v > best) { best = v; bidx = besti[row * 32 + sl]; }
        }
        long long gm = rowbase + row;
        float lsum = 0.0f;
#pragma unroll
        for (int l = 0; l < LAT; ++l) {
            float q  = codebook[l * NEMB + bidx];
            float zv = ob[row * OUT3P + l];
            out[(long long)ZQ_OFF + gm * LAT + l] = q;   // z_q_st forward value
            float d = q - zv;
            lsum += d * d;
        }
        out[SCALER_OFF + gm]   = ob[row * OUT3P + 16];
        out[REDSHIFT_OFF + gm] = ob[row * OUT3P + 17];
        out[IDS_OFF + gm]      = (float)bidx;
        lossp[row] = lsum;
    }
    __syncthreads();
    if (tid == 0) {
        float s = 0.0f;
        for (int i = 0; i < BM; ++i) s += lossp[i];
        // mean((zq-zl)^2) * (1 + BETA)
        atomicAdd(&out[LOSS_OFF], s * (1.25f / ((float)N_ROWS * (float)LAT)));
    }
}

// ---------------------------------------------------------------------------
extern "C" void kernel_launch(void* const* d_in, const int* in_sizes, int n_in,
                              void* d_out, int out_size, void* d_ws, size_t ws_size,
                              hipStream_t stream)
{
    (void)in_sizes; (void)n_in; (void)out_size; (void)ws_size;

    const float* z   = (const float*)d_in[0];
    const float* W0  = (const float*)d_in[1];
    const float* b0  = (const float*)d_in[2];
    const float* W1  = (const float*)d_in[3];
    const float* b1  = (const float*)d_in[4];
    const float* W2  = (const float*)d_in[5];
    const float* b2  = (const float*)d_in[6];
    const float* W3  = (const float*)d_in[7];
    const float* b3  = (const float*)d_in[8];
    const float* cb  = (const float*)d_in[9];

    _Float16* ws  = (_Float16*)d_ws;
    _Float16* W0p = ws + W0P_OFF;
    _Float16* W1p = ws + W1P_OFF;
    _Float16* W2p = ws + W2P_OFF;
    _Float16* W3p = ws + W3P_OFF;
    _Float16* CBp = ws + CBP_OFF;
    float*    cb2 = (float*)(ws + CB2_OFF_H);
    float*    out = (float*)d_out;

    // weight pre-pack (f32 -> f16 WMMA-B fragment layout, L2-resident)
    pack_b_kernel<<<(W0P_SZ + 255) / 256, 256, 0, stream>>>(W0, W0p, 32, 32, HID, HID);
    pack_b_kernel<<<(W1P_SZ + 255) / 256, 256, 0, stream>>>(W1, W1p, HID, HID, HID, HID);
    pack_b_kernel<<<(W2P_SZ + 255) / 256, 256, 0, stream>>>(W2, W2p, HID, HID, HID, HID);
    pack_b_kernel<<<(W3P_SZ + 255) / 256, 256, 0, stream>>>(W3, W3p, HID, HID, OUT3P, OUT3A);
    pack_b_kernel<<<(CBP_SZ + 255) / 256, 256, 0, stream>>>(cb, CBp, 32, LAT, NEMB, NEMB);
    cb2_kernel<<<(NEMB + 255) / 256, 256, 0, stream>>>(cb, cb2);
    zero_loss_kernel<<<1, 32, 0, stream>>>(out);

    const size_t smem_bytes =
        (size_t)(BM * 32 + 2 * BM * HID) * sizeof(_Float16) +   // zb + hA + hB
        (size_t)(BM * OUT3P + BM * 32 + BM * 32 + BM) * 4;      // ob + bestv + besti + lossp

    fused_decoder_vq<<<N_ROWS / BM, THREADS, smem_bytes, stream>>>(
        z, W0p, b0, W1p, b1, W2p, b2, W3p, b3, CBp, cb2, cb, out);
}
